// EmbeddingAttention_42889543418181
// MI455X (gfx1250) — matrix operand order repaired and checked by reading, not compiled, not run on hardware
//
#include <hip/hip_runtime.h>
#include <hip/hip_bf16.h>

// ---------------------------------------------------------------------------
// GAT-style attention, fused for MI455X (gfx1250, wave32)
//   f1 = emb @ (W a1), f2 = emb @ (W a2)          (rank-structured scores)
//   attn = softmax_j(leaky_relu(f1[i]+f2[j]))     (regenerated on the fly)
//   out[b'] = attn[b'%32] @ x[b']                 (bf16 WMMA 16x16x32, f32 acc)
// x tiles move global->LDS via GLOBAL_LOAD_ASYNC_TO_LDS_B128 (ASYNCcnt),
// then get converted to a transposed bf16 tile laid out as packed WMMA-B
// K-pairs (stride 36 halves -> conflict-free b32 reads).
// ---------------------------------------------------------------------------

typedef __attribute__((ext_vector_type(16))) __bf16 v16bf;
typedef __attribute__((ext_vector_type(8)))  float  v8f;
typedef int v4i __attribute__((__vector_size__(4 * sizeof(int))));
typedef __attribute__((address_space(1))) v4i* gptr_b128;
typedef __attribute__((address_space(3))) v4i* lptr_b128;

#define ALPHA 0.2f

#if defined(__has_builtin)
# if __has_builtin(__builtin_amdgcn_global_load_async_to_lds_b128)
#  define HAVE_ASYNC 1
# endif
#endif
#ifndef HAVE_ASYNC
# define HAVE_ASYNC 0
#endif

__device__ __forceinline__ void wait_async_all() {
#if HAVE_ASYNC
# if __has_builtin(__builtin_amdgcn_s_wait_asynccnt)
    __builtin_amdgcn_s_wait_asynccnt(0);
# else
    asm volatile("s_wait_asynccnt 0" ::: "memory");
# endif
#endif
}

// pack two floats into one dword of 2 x bf16 (lo = a, hi = b)
// round-to-nearest (+0x8000) then one v_perm_b32 splices the two high halves.
__device__ __forceinline__ unsigned int pkbf(float a, float b) {
#if defined(__has_builtin) && __has_builtin(__builtin_amdgcn_cvt_pk_bf16_f32)
    auto t = __builtin_amdgcn_cvt_pk_bf16_f32(a, b);   // v_cvt_pk_bf16_f32
    unsigned int u;
    __builtin_memcpy(&u, &t, 4);
    return u;
#else
    const unsigned int ua = __float_as_uint(a) + 0x8000u;
    const unsigned int ub = __float_as_uint(b) + 0x8000u;
    // concat {ub,ua}: bytes 2,3 of ua -> dst bytes 0,1 ; bytes 2,3 of ub -> 2,3
    return __builtin_amdgcn_perm(ub, ua, 0x07060302u); // v_perm_b32
#endif
}

__device__ __forceinline__ float lrelu(float s) {
    return fmaxf(s, ALPHA * s);
}

// ---------------------------------------------------------------------------
// Kernel 1: f1[b,t] = emb[b,t,:].(W a1),  f2[b,t] = emb[b,t,:].(W a2)
// ---------------------------------------------------------------------------
__global__ __launch_bounds__(256)
void gat_f12_kernel(const float* __restrict__ emb,
                    const float* __restrict__ W,
                    const float* __restrict__ a,
                    float* __restrict__ f1,
                    float* __restrict__ f2) {
    __shared__ float wa1[64];
    __shared__ float wa2[64];
    const int tid = threadIdx.x;
    if (tid < 64) {
        float s1 = 0.f, s2 = 0.f;
        #pragma unroll 8
        for (int h = 0; h < 64; ++h) {
            const float w = W[tid * 64 + h];
            s1 += w * a[h];
            s2 += w * a[64 + h];
        }
        wa1[tid] = s1;
        wa2[tid] = s2;
    }
    __syncthreads();

    const int row = blockIdx.x * 256 + tid;            // 0..32767
    const float4* e4 = (const float4*)(emb + (size_t)row * 64);
    float s1 = 0.f, s2 = 0.f;
    #pragma unroll
    for (int q = 0; q < 16; ++q) {
        const float4 v = e4[q];
        s1 += v.x * wa1[4*q+0] + v.y * wa1[4*q+1] + v.z * wa1[4*q+2] + v.w * wa1[4*q+3];
        s2 += v.x * wa2[4*q+0] + v.y * wa2[4*q+1] + v.z * wa2[4*q+2] + v.w * wa2[4*q+3];
    }
    f1[row] = s1;
    f2[row] = s2;
}

// ---------------------------------------------------------------------------
// Kernel 2: maxf2[b] = max_t f2[b,t]
// ---------------------------------------------------------------------------
__global__ __launch_bounds__(256)
void gat_max_kernel(const float* __restrict__ f2, float* __restrict__ maxf2) {
    __shared__ float red[256];
    const int b = blockIdx.x, tid = threadIdx.x;
    float m = -3.4e38f;
    #pragma unroll
    for (int q = 0; q < 4; ++q)
        m = fmaxf(m, f2[b * 1024 + tid + q * 256]);
    red[tid] = m;
    __syncthreads();
    #pragma unroll
    for (int s = 128; s > 0; s >>= 1) {
        if (tid < s) red[tid] = fmaxf(red[tid], red[tid + s]);
        __syncthreads();
    }
    if (tid == 0) maxf2[b] = red[0];
}

// ---------------------------------------------------------------------------
// Kernel 3: Zi[b,i] = 1 / sum_j exp(lrelu(f1+f2) - M)
// ---------------------------------------------------------------------------
__global__ __launch_bounds__(256)
void gat_denom_kernel(const float* __restrict__ f1,
                      const float* __restrict__ f2,
                      const float* __restrict__ maxf2,
                      float* __restrict__ Zi) {
    __shared__ float f2s[1024];
    const int b = blockIdx.x >> 2, q = blockIdx.x & 3, tid = threadIdx.x;
    #pragma unroll
    for (int t = 0; t < 4; ++t)
        f2s[tid + t * 256] = f2[b * 1024 + tid + t * 256];
    __syncthreads();

    const int   i   = q * 256 + tid;
    const float f1v = f1[b * 1024 + i];
    const float Mxv = lrelu(f1v + maxf2[b]);
    float sum = 0.f;
    #pragma unroll 4
    for (int j = 0; j < 1024; ++j)
        sum += __expf(lrelu(f1v + f2s[j]) - Mxv);
    Zi[b * 1024 + i] = 1.0f / sum;
}

// ---------------------------------------------------------------------------
// Kernel 4 (main): out[b'] = softmax(e[b'%32]) @ x[b']
// grid = 64*8 (b' x 128-row chunk), block = 256 (8 waves).
// Each wave: one 16-row M-tile, N=64 (4 WMMA N-tiles), 32 K-steps of 32.
// Pipeline per K-step: async DMA (k+1 -> raw LDS) overlaps WMMA compute (k);
// tail: wait_asynccnt + convert raw(k+1) -> transposed bf16 xT(k+1).
// ---------------------------------------------------------------------------
#define XT_STRIDE 36   // halves per column; 72B stride -> conflict-free b32 reads

__global__ __launch_bounds__(256)
void gat_main_kernel(const float* __restrict__ x,
                     const float* __restrict__ f1,
                     const float* __restrict__ f2,
                     const float* __restrict__ maxf2,
                     const float* __restrict__ Zi,
                     float* __restrict__ out) {
    __shared__ float          f2s[1024];
    __shared__ float          xRaw[2][32 * 64];        // raw async landing pad
    __shared__ unsigned short xT[2][64 * XT_STRIDE];   // [col*36 + j], bf16 bits
    __shared__ float          f1S[128];
    __shared__ float          ZiS[128];

    const int bp    = blockIdx.x >> 3;     // 0..63  (output batch)
    const int chunk = blockIdx.x & 7;      // 0..7   (128-row chunk)
    const int b     = bp & 31;             // attn batch (tile along axis 0)
    const int i0    = chunk * 128;

    const int tid  = threadIdx.x;
    const int wave = tid >> 5;
    const int lane = tid & 31;
    const int n    = lane & 15;            // A: M row ; B/C: N col
    const int hf   = lane >> 4;            // half-wave selector

    const float* xb   = x   + (size_t)bp * 1024 * 64;
    float*       outb = out + (size_t)bp * 1024 * 64;

    // ---- stage f2 row (4KB) and per-row stats for this chunk ----
    #pragma unroll
    for (int q = 0; q < 4; ++q)
        f2s[tid + q * 256] = f2[b * 1024 + tid + q * 256];
    if (tid < 128) {
        f1S[tid] = f1[b * 1024 + i0 + tid];
        ZiS[tid] = Zi[b * 1024 + i0 + tid];
    }

    // ---- async stage: 32 rows(j) x 64 f32 (8KB) -> xRaw[buf], contiguous ----
    auto stageAsync = [&](int buf, int j0) {
        #pragma unroll
        for (int it = 0; it < 2; ++it) {
            const int idx = tid + it * 256;            // 0..511 b128 chunks
            const int j   = idx >> 4;
            const int c4  = idx & 15;
            const float* g = xb + (size_t)(j0 + j) * 64 + c4 * 4;
            float*       l = &xRaw[buf][j * 64 + c4 * 4];
#if HAVE_ASYNC
            __builtin_amdgcn_global_load_async_to_lds_b128(
                (gptr_b128)g, (lptr_b128)l, 0, 0);
#else
            *(float4*)l = *(const float4*)g;
#endif
        }
    };

    // ---- convert raw f32 tile -> transposed bf16 tile (packed K-pairs) ----
    // thread owns rows (jb, jb+1) at columns c4*4..c4*4+3 -> 4 packed b32 stores
    auto convertTile = [&](int buf) {
        const int c4 = tid & 15;
        const int jb = (tid >> 4) << 1;                // even row base 0..30
        const float4 v0 = *(const float4*)&xRaw[buf][(jb + 0) * 64 + c4 * 4];
        const float4 v1 = *(const float4*)&xRaw[buf][(jb + 1) * 64 + c4 * 4];
        *(unsigned int*)&xT[buf][(c4 * 4 + 0) * XT_STRIDE + jb] = pkbf(v0.x, v1.x);
        *(unsigned int*)&xT[buf][(c4 * 4 + 1) * XT_STRIDE + jb] = pkbf(v0.y, v1.y);
        *(unsigned int*)&xT[buf][(c4 * 4 + 2) * XT_STRIDE + jb] = pkbf(v0.z, v1.z);
        *(unsigned int*)&xT[buf][(c4 * 4 + 3) * XT_STRIDE + jb] = pkbf(v0.w, v1.w);
    };

    // ---- pipeline preamble ----
    stageAsync(0, 0);
    wait_async_all();
    __syncthreads();
    convertTile(0);
    stageAsync(1, 32);         // in flight during first compute step
    __syncthreads();

    const float f1v = f1S[wave * 16 + n];
    const float Mxv = lrelu(f1v + maxf2[b]);

    v8f acc[4] = {};

    for (int ks = 0; ks < 32; ++ks) {
        const int buf = ks & 1;
        const int j0  = ks * 32;

        // ---- A fragment: P(i,j) = exp(lrelu(f1+f2) - M), WMMA 16-bit layout
        //      VGPR v holds K = 16*(v>>2) + 8*hf + 2*(v&3) + {0,1}
        union { v16bf v; unsigned int u[8]; } A;
        #pragma unroll
        for (int v = 0; v < 8; ++v) {
            const int jj = ((v >> 2) << 4) + (hf << 3) + ((v & 3) << 1);
            const float s0 = f1v + f2s[j0 + jj];
            const float s1 = f1v + f2s[j0 + jj + 1];
            const float e0 = __expf(lrelu(s0) - Mxv);
            const float e1 = __expf(lrelu(s1) - Mxv);
            A.u[v] = pkbf(e0, e1);
        }

        // ---- 4 N-tiles: B from LDS (packed K-pairs), then WMMA ----
        #pragma unroll
        for (int nt = 0; nt < 4; ++nt) {
            const int col = nt * 16 + n;
            union { v16bf v; unsigned int u[8]; } B;
            #pragma unroll
            for (int v = 0; v < 8; ++v) {
                const int jj = ((v >> 2) << 4) + (hf << 3) + ((v & 3) << 1);   // K = 16*(v>>2) + 8*hf + 2*(v&3) {,+1}
                B.u[v] = *(const unsigned int*)&xT[buf][col * XT_STRIDE + jj];
            }
            acc[nt] = __builtin_amdgcn_wmma_f32_16x16x32_bf16(
                false, A.v, false, B.v, (short)0, acc[nt], false, false);
        }

        // ---- pipeline tail: land async tile (ks+1), convert it, refill ----
        if (ks + 1 < 32) {
            wait_async_all();          // xRaw[buf^1] landed
            __syncthreads();           // everyone done with xT[buf^1] readers
            convertTile(buf ^ 1);      // xRaw[buf^1] -> xT[buf^1]
            if (ks + 2 < 32)
                stageAsync(buf, j0 + 64);   // refill raw[buf] for ks+2
            __syncthreads();           // xT[buf^1] visible for next step
        }
    }

    // ---- C tile: lane holds col n, rows r + 8*hf ; apply 1/Z row scale ----
    #pragma unroll
    for (int nt = 0; nt < 4; ++nt) {
        #pragma unroll
        for (int r = 0; r < 8; ++r) {
            const int il = wave * 16 + hf * 8 + r;
            outb[(size_t)(i0 + il) * 64 + nt * 16 + n] = acc[nt][r] * ZiS[il];
        }
    }
}

// ---------------------------------------------------------------------------
extern "C" void kernel_launch(void* const* d_in, const int* in_sizes, int n_in,
                              void* d_out, int out_size, void* d_ws, size_t ws_size,
                              hipStream_t stream) {
    (void)in_sizes; (void)n_in; (void)out_size; (void)ws_size;

    const float* emb = (const float*)d_in[0];   // (32,1024,64)
    const float* x   = (const float*)d_in[1];   // (64,1024,64)
    const float* W   = (const float*)d_in[2];   // (64,64)
    const float* a   = (const float*)d_in[3];   // (128,1)
    float*       out = (float*)d_out;           // (64,1024,64)

    float* ws   = (float*)d_ws;
    float* f1   = ws;                 // 32768
    float* f2   = ws + 32768;         // 32768
    float* mxf2 = ws + 65536;         // 32
    float* Zi   = ws + 65536 + 32;    // 32768   (~393 KB total)

    gat_f12_kernel  <<<128, 256, 0, stream>>>(emb, W, a, f1, f2);
    gat_max_kernel  <<<32,  256, 0, stream>>>(f2, mxf2);
    gat_denom_kernel<<<128, 256, 0, stream>>>(f1, f2, mxf2, Zi);
    gat_main_kernel <<<512, 256, 0, stream>>>(x, f1, f2, mxf2, Zi, out);
}
